// CausalSelfAttention_54348516163806
// MI455X (gfx1250) — matrix operand order repaired and checked by reading, not compiled, not run on hardware
//
#include <hip/hip_runtime.h>
#include <hip/hip_bf16.h>

// ---------------------------------------------------------------------------
// CausalSelfAttention for MI455X (gfx1250, wave32, WMMA).
// B=4, T=1024, C=1024, H=16, d=64.
// f32 inputs are converted to bf16 while staging into LDS; all matmuls run on
// v_wmma_f32_16x16x32_bf16 (f32 accumulate). LDS tiles are laid out so every
// WMMA fragment read is contiguous (A row-major, B transposed) -> ds_load_b128.
// GEMMs stage K=64 panels (16 WMMAs per barrier round). Async global->LDS DMA
// (gfx1250) is used for pure bf16 tile copies.
// ---------------------------------------------------------------------------

typedef __attribute__((ext_vector_type(16))) __bf16 v16bf;
typedef __attribute__((ext_vector_type(8)))  __bf16 v8bf;
typedef __attribute__((ext_vector_type(4)))  __bf16 v4bf;
typedef __attribute__((ext_vector_type(8)))  float  v8f;
typedef __attribute__((ext_vector_type(4)))  float  v4f;
typedef __attribute__((ext_vector_type(4)))  int    v4i;

#if defined(__gfx1250__) &&                                        \
    __has_builtin(__builtin_amdgcn_global_load_async_to_lds_b128) && \
    __has_builtin(__builtin_amdgcn_s_wait_asynccnt)
#define USE_ASYNC_LDS 1
#else
#define USE_ASYNC_LDS 0
#endif

#if USE_ASYNC_LDS
__device__ __forceinline__ void async_copy_b128(const void* g, void* l) {
  // signature (from hipcc diagnostic): (v4i __device__*, v4i __shared__*, Ii, Ii)
  __builtin_amdgcn_global_load_async_to_lds_b128(
      (__attribute__((address_space(1))) v4i*)g,
      (__attribute__((address_space(3))) v4i*)l, 0, 0);
}
__device__ __forceinline__ void async_wait0() {
  __builtin_amdgcn_s_wait_asynccnt(0);
}
#endif

__device__ __forceinline__ v8f wmma_bf16(v16bf a, v16bf b, v8f c) {
  // 8 args: (neg_a, A, neg_b, B, c_mod, C, reuse_a, reuse_b)
  return __builtin_amdgcn_wmma_f32_16x16x32_bf16(false, a, false, b, (short)0, c,
                                                 false, false);
}

constexpr int Bn = 4;
constexpr int Tn = 1024;
constexpr int Cn = 1024;
constexpr int Hn = 16;
constexpr int Dn = 64;           // head dim
constexpr int Mtot = Bn * Tn;    // 4096 rows

// ---------------------------------------------------------------------------
// Kernel 1: qkv = x @ w_attn + b_attn, scattered into q/k/v [B,H,T,64] bf16.
// M=4096, N=3072, K=1024. WG = 256 threads (8 waves), tile 128x128, K-step 64.
// As: [row][k] 128x64 (+pad -> 160B stride, 16B aligned).
// Bst: [col][k] 128x64 (TRANSPOSED so B-fragment reads are contiguous).
// 16 WMMAs per wave per barrier round; 16 rounds total.
// ---------------------------------------------------------------------------
__global__ __launch_bounds__(256)
void qkv_gemm(const float* __restrict__ x, const float* __restrict__ w,
              const float* __restrict__ bias,
              __bf16* __restrict__ q, __bf16* __restrict__ k,
              __bf16* __restrict__ v) {
  __shared__ __bf16 As[128][80];    // 128 x 64 (+16 pad)
  __shared__ __bf16 Bst[128][80];   // 128 cols x 64 k (+16 pad), transposed

  const int lane  = threadIdx.x & 31;
  const int wave  = threadIdx.x >> 5;   // 0..7
  const int waveM = wave >> 1;          // 0..3 -> 32-row strips
  const int waveN = wave & 1;           // 0..1 -> 64-col strips
  const int tileM = blockIdx.x * 128;
  const int tileN = blockIdx.y * 128;

  v8f acc[2][4] = {};

  const int am  = lane & 15;
  const int akb = (lane < 16) ? 0 : 16;   // A-frag K base (contiguous k = 16h + s)
  const int bn  = lane & 15;
  const int bkb = (lane < 16) ? 0 : 16;   // B-frag K base

  for (int k0 = 0; k0 < Cn; k0 += 64) {
    if (k0 + 64 < Cn) {  // prefetch next K panel -> global_prefetch_b8
      __builtin_prefetch(&x[(size_t)(tileM + (threadIdx.x >> 1)) * Cn + k0 + 64 +
                            (threadIdx.x & 1) * 32], 0, 0);
      __builtin_prefetch(&w[(size_t)(k0 + 64 + (threadIdx.x >> 2)) * (3 * Cn) +
                            tileN + (threadIdx.x & 3) * 32], 0, 0);
    }
    // Stage A: 128x64 f32 -> bf16, float4 loads, b64 LDS stores
    for (int i = threadIdx.x; i < 128 * 16; i += 256) {
      int r = i >> 4, c = (i & 15) * 4;
      v4f f = *(const v4f*)&x[(size_t)(tileM + r) * Cn + k0 + c];
      v4bf h;
      h[0] = (__bf16)f[0]; h[1] = (__bf16)f[1];
      h[2] = (__bf16)f[2]; h[3] = (__bf16)f[3];
      *(v4bf*)&As[r][c] = h;
    }
    // Stage B transposed: float4 along N, 4 scalar b16 scatter stores
    for (int i = threadIdx.x; i < 64 * 32; i += 256) {
      int r = i >> 5, c = (i & 31) * 4;
      v4f f = *(const v4f*)&w[(size_t)(k0 + r) * (3 * Cn) + tileN + c];
      Bst[c + 0][r] = (__bf16)f[0];
      Bst[c + 1][r] = (__bf16)f[1];
      Bst[c + 2][r] = (__bf16)f[2];
      Bst[c + 3][r] = (__bf16)f[3];
    }
    __syncthreads();

#pragma unroll
    for (int kc = 0; kc < 2; ++kc) {   // two K=32 chunks from the staged panel
      v16bf afrag[2];
#pragma unroll
      for (int i = 0; i < 2; ++i) {
        int row = waveM * 32 + i * 16 + am;
#pragma unroll
        for (int e = 0; e < 8; ++e) {
          afrag[i][e]     = As[row][kc * 32 + akb + e];
          afrag[i][e + 8] = As[row][kc * 32 + akb + 8 + e];
        }
      }
      v16bf bfrag[4];
#pragma unroll
      for (int j = 0; j < 4; ++j) {
        int col = waveN * 64 + j * 16 + bn;
#pragma unroll
        for (int e = 0; e < 16; ++e) bfrag[j][e] = Bst[col][kc * 32 + bkb + e];
      }
#pragma unroll
      for (int i = 0; i < 2; ++i)
#pragma unroll
        for (int j = 0; j < 4; ++j)
          acc[i][j] = wmma_bf16(afrag[i], bfrag[j], acc[i][j]);
    }
    __syncthreads();
  }

  // Epilogue: bias + scatter into q/k/v [B,H,T,64]
  const int cn = lane & 15;
  const int mb = (lane < 16) ? 0 : 8;
#pragma unroll
  for (int i = 0; i < 2; ++i) {
#pragma unroll
    for (int j = 0; j < 4; ++j) {
      int ncol   = tileN + waveN * 64 + j * 16 + cn;
      float bval = bias[ncol];
      int which  = ncol >> 10;          // 0=q 1=k 2=v
      int c      = ncol & (Cn - 1);
      int h      = c >> 6;
      int d      = c & 63;
      __bf16* dst = (which == 0) ? q : (which == 1) ? k : v;
#pragma unroll
      for (int r = 0; r < 8; ++r) {
        int mrow = tileM + waveM * 32 + i * 16 + mb + r;
        int b    = mrow >> 10;
        int t    = mrow & (Tn - 1);
        float val = acc[i][j][r] + bval;
        dst[((size_t)(b * Hn + h) * Tn + t) * Dn + d] = (__bf16)val;
      }
    }
  }
}

// ---------------------------------------------------------------------------
// Kernel 2: flash attention. grid = (T/64, B*H). WG = 128 threads (4 waves),
// each wave owns 16 query rows. Online softmax over key tiles of 64.
// Qs/Ks: [row][d] (fragment reads contiguous). Vst: [d][key] (transposed so
// the P@V B-fragment reads are contiguous). Row stride 72 elems = 144B,
// 16B-aligned and conflict-free across 16 lanes.
// ---------------------------------------------------------------------------
__global__ __launch_bounds__(128)
void flash_attn(const __bf16* __restrict__ Q, const __bf16* __restrict__ K,
                const __bf16* __restrict__ V, __bf16* __restrict__ Y) {
  __shared__ __bf16 Qs[64][72];
  __shared__ __bf16 Ks[64][72];
  __shared__ __bf16 Vst[64][72];     // [d][key]
  __shared__ __bf16 Ps[4][16][72];   // per-wave P tile (C-layout -> A-layout)

  const int lane = threadIdx.x & 31;
  const int wave = threadIdx.x >> 5;      // 0..3
  const int qt   = blockIdx.x;            // query tile (64 rows)
  const int bh   = blockIdx.y;            // b*H + h
  const size_t base = (size_t)bh * Tn * Dn;
  const __bf16* qp = Q + base;
  const __bf16* kp = K + base;
  const __bf16* vp = V + base;

  // Stage Q tile once (pure bf16 copy -> async DMA path when available)
#if USE_ASYNC_LDS
  for (int i = threadIdx.x; i < 64 * 8; i += 128) {
    int r = i >> 3, c = (i & 7) * 8;
    async_copy_b128(&qp[(size_t)(qt * 64 + r) * Dn + c], &Qs[r][c]);
  }
  async_wait0();
#else
  for (int i = threadIdx.x; i < 64 * 8; i += 128) {
    int r = i >> 3, c = (i & 7) * 8;
    *(v8bf*)&Qs[r][c] = *(const v8bf*)&qp[(size_t)(qt * 64 + r) * Dn + c];
  }
#endif
  __syncthreads();

  const int am  = lane & 15;
  const int akb = (lane < 16) ? 0 : 16;
  const int bn  = lane & 15;
  const int bkb = (lane < 16) ? 0 : 16;
  const int hi  = (lane < 16) ? 0 : 8;    // C/D row base

  v16bf qfrag[2];
#pragma unroll
  for (int kc = 0; kc < 2; ++kc)
#pragma unroll
    for (int e = 0; e < 8; ++e) {
      qfrag[kc][e]     = Qs[wave * 16 + am][kc * 32 + akb + e];
      qfrag[kc][e + 8] = Qs[wave * 16 + am][kc * 32 + akb + 8 + e];
    }

  float m_i[8], l_i[8];
#pragma unroll
  for (int r = 0; r < 8; ++r) { m_i[r] = -1e30f; l_i[r] = 0.0f; }
  v8f o[4] = {};
  const float scale = 0.125f;  // 1/sqrt(64)

  for (int kt = 0; kt <= qt; ++kt) {
    __syncthreads();   // previous Ks/Vst readers done
    // K tile: straight copy [key][d]
#if USE_ASYNC_LDS
    for (int i = threadIdx.x; i < 64 * 8; i += 128) {
      int r = i >> 3, c = (i & 7) * 8;
      async_copy_b128(&kp[(size_t)(kt * 64 + r) * Dn + c], &Ks[r][c]);
    }
#else
    for (int i = threadIdx.x; i < 64 * 8; i += 128) {
      int r = i >> 3, c = (i & 7) * 8;
      *(v8bf*)&Ks[r][c] = *(const v8bf*)&kp[(size_t)(kt * 64 + r) * Dn + c];
    }
#endif
    // V tile: transpose into Vst[d][key] (vector global read, b16 scatter)
    for (int i = threadIdx.x; i < 64 * 8; i += 128) {
      int r = i >> 3, c = (i & 7) * 8;
      v8bf t = *(const v8bf*)&vp[(size_t)(kt * 64 + r) * Dn + c];
#pragma unroll
      for (int e = 0; e < 8; ++e) Vst[c + e][r] = t[e];
    }
#if USE_ASYNC_LDS
    async_wait0();
#endif
    __syncthreads();

    // S = Q @ K^T  (4 key sub-tiles of 16); B[kd][n] = K[n][kd] contiguous
    v8f s[4] = {};
#pragma unroll
    for (int j = 0; j < 4; ++j) {
#pragma unroll
      for (int kc = 0; kc < 2; ++kc) {
        v16bf bfrag;
#pragma unroll
        for (int e = 0; e < 16; ++e)
          bfrag[e] = Ks[j * 16 + bn][kc * 32 + bkb + e];
        s[j] = wmma_bf16(qfrag[kc], bfrag, s[j]);
      }
    }

    // Scale + causal mask
#pragma unroll
    for (int j = 0; j < 4; ++j) {
      int key = kt * 64 + j * 16 + (lane & 15);
#pragma unroll
      for (int r = 0; r < 8; ++r) {
        int qrow = qt * 64 + wave * 16 + hi + r;
        float sv = s[j][r] * scale;
        if (key > qrow) sv = -1e30f;
        s[j][r] = sv;
      }
    }

    // Row max across 4 tiles, then across 16 lanes (xor masks stay in half)
    float rmax[8];
#pragma unroll
    for (int r = 0; r < 8; ++r) {
      float mx = s[0][r];
#pragma unroll
      for (int j = 1; j < 4; ++j) mx = fmaxf(mx, s[j][r]);
#pragma unroll
      for (int off = 1; off < 16; off <<= 1)
        mx = fmaxf(mx, __shfl_xor(mx, off, 32));
      rmax[r] = mx;
    }

    float m_new[8], alpha[8], rsum[8];
#pragma unroll
    for (int r = 0; r < 8; ++r) {
      m_new[r] = fmaxf(m_i[r], rmax[r]);
      alpha[r] = __expf(m_i[r] - m_new[r]);
      rsum[r]  = 0.0f;
    }

    // P = exp(S - m_new); route through per-wave LDS to A-fragment layout
#pragma unroll
    for (int j = 0; j < 4; ++j) {
#pragma unroll
      for (int r = 0; r < 8; ++r) {
        float p = __expf(s[j][r] - m_new[r]);
        rsum[r] += p;
        Ps[wave][hi + r][j * 16 + (lane & 15)] = (__bf16)p;
      }
    }
#pragma unroll
    for (int r = 0; r < 8; ++r) {
#pragma unroll
      for (int off = 1; off < 16; off <<= 1)
        rsum[r] += __shfl_xor(rsum[r], off, 32);
      l_i[r] = l_i[r] * alpha[r] + rsum[r];
      m_i[r] = m_new[r];
    }
#pragma unroll
    for (int j = 0; j < 4; ++j)
#pragma unroll
      for (int r = 0; r < 8; ++r) o[j][r] *= alpha[r];

    __builtin_amdgcn_wave_barrier();  // order per-wave LDS P writes vs reads

    // O += P @ V: A = P[m][key] (contiguous), B = Vst[d][key] (contiguous)
#pragma unroll
    for (int j = 0; j < 4; ++j) {       // d sub-tile
#pragma unroll
      for (int kc = 0; kc < 2; ++kc) {  // 32-key chunk
        v16bf pfrag, vfrag;
#pragma unroll
        for (int e = 0; e < 8; ++e) {
          pfrag[e]     = Ps[wave][am][kc * 32 + akb + e];
          pfrag[e + 8] = Ps[wave][am][kc * 32 + akb + 8 + e];
        }
#pragma unroll
        for (int e = 0; e < 16; ++e)
          vfrag[e] = Vst[j * 16 + bn][kc * 32 + bkb + e];
        o[j] = wmma_bf16(pfrag, vfrag, o[j]);
      }
    }
  }

  // Finalize: O /= l_i, write y[b,t,h*64+d] as bf16
  const int b = bh >> 4;
  const int h = bh & 15;
#pragma unroll
  for (int j = 0; j < 4; ++j) {
    int d = j * 16 + (lane & 15);
#pragma unroll
    for (int r = 0; r < 8; ++r) {
      int t = qt * 64 + wave * 16 + hi + r;
      float val = o[j][r] / l_i[r];
      Y[((size_t)(b)*Tn + t) * Cn + h * Dn + d] = (__bf16)val;
    }
  }
}

// ---------------------------------------------------------------------------
// Kernel 3: out = y @ w_proj + b_proj. M=4096, N=1024, K=1024, K-step 64.
// A (y) is already bf16 -> async DMA staging; B converted+transposed; f32 out.
// ---------------------------------------------------------------------------
__global__ __launch_bounds__(256)
void proj_gemm(const __bf16* __restrict__ y, const float* __restrict__ w,
               const float* __restrict__ bias, float* __restrict__ out) {
  __shared__ __bf16 As[128][80];
  __shared__ __bf16 Bst[128][80];   // transposed [col][k]

  const int lane  = threadIdx.x & 31;
  const int wave  = threadIdx.x >> 5;
  const int waveM = wave >> 1;
  const int waveN = wave & 1;
  const int tileM = blockIdx.x * 128;
  const int tileN = blockIdx.y * 128;

  v8f acc[2][4] = {};
  const int am  = lane & 15;
  const int akb = (lane < 16) ? 0 : 16;
  const int bn  = lane & 15;
  const int bkb = (lane < 16) ? 0 : 16;

  for (int k0 = 0; k0 < Cn; k0 += 64) {
    if (k0 + 64 < Cn) {
      __builtin_prefetch(&y[(size_t)(tileM + (threadIdx.x >> 1)) * Cn + k0 + 64 +
                            (threadIdx.x & 1) * 32], 0, 0);
      __builtin_prefetch(&w[(size_t)(k0 + 64 + (threadIdx.x >> 2)) * Cn + tileN +
                            (threadIdx.x & 3) * 32], 0, 0);
    }
    // Stage A (bf16 -> bf16): async DMA when available
#if USE_ASYNC_LDS
    for (int i = threadIdx.x; i < 128 * 8; i += 256) {
      int r = i >> 3, c = (i & 7) * 8;
      async_copy_b128(&y[(size_t)(tileM + r) * Cn + k0 + c], &As[r][c]);
    }
#else
    for (int i = threadIdx.x; i < 128 * 8; i += 256) {
      int r = i >> 3, c = (i & 7) * 8;
      *(v8bf*)&As[r][c] = *(const v8bf*)&y[(size_t)(tileM + r) * Cn + k0 + c];
    }
#endif
    // Stage B transposed (f32 -> bf16)
    for (int i = threadIdx.x; i < 64 * 32; i += 256) {
      int r = i >> 5, c = (i & 31) * 4;
      v4f f = *(const v4f*)&w[(size_t)(k0 + r) * Cn + tileN + c];
      Bst[c + 0][r] = (__bf16)f[0];
      Bst[c + 1][r] = (__bf16)f[1];
      Bst[c + 2][r] = (__bf16)f[2];
      Bst[c + 3][r] = (__bf16)f[3];
    }
#if USE_ASYNC_LDS
    async_wait0();
#endif
    __syncthreads();

#pragma unroll
    for (int kc = 0; kc < 2; ++kc) {
      v16bf afrag[2];
#pragma unroll
      for (int i = 0; i < 2; ++i) {
        int row = waveM * 32 + i * 16 + am;
#pragma unroll
        for (int e = 0; e < 8; ++e) {
          afrag[i][e]     = As[row][kc * 32 + akb + e];
          afrag[i][e + 8] = As[row][kc * 32 + akb + 8 + e];
        }
      }
      v16bf bfrag[4];
#pragma unroll
      for (int j = 0; j < 4; ++j) {
        int col = waveN * 64 + j * 16 + bn;
#pragma unroll
        for (int e = 0; e < 16; ++e) bfrag[j][e] = Bst[col][kc * 32 + bkb + e];
      }
#pragma unroll
      for (int i = 0; i < 2; ++i)
#pragma unroll
        for (int j = 0; j < 4; ++j)
          acc[i][j] = wmma_bf16(afrag[i], bfrag[j], acc[i][j]);
    }
    __syncthreads();
  }

  const int cn = lane & 15;
  const int mb = (lane < 16) ? 0 : 8;
#pragma unroll
  for (int i = 0; i < 2; ++i)
#pragma unroll
    for (int j = 0; j < 4; ++j) {
      int ncol   = tileN + waveN * 64 + j * 16 + cn;
      float bval = bias[ncol];
#pragma unroll
      for (int r = 0; r < 8; ++r) {
        int mrow = tileM + waveM * 32 + i * 16 + mb + r;
        out[(size_t)mrow * Cn + ncol] = acc[i][j][r] + bval;
      }
    }
}

// ---------------------------------------------------------------------------
extern "C" void kernel_launch(void* const* d_in, const int* in_sizes, int n_in,
                              void* d_out, int out_size, void* d_ws,
                              size_t ws_size, hipStream_t stream) {
  const float* x      = (const float*)d_in[0];
  const float* w_attn = (const float*)d_in[1];
  const float* b_attn = (const float*)d_in[2];
  const float* w_proj = (const float*)d_in[3];
  const float* b_proj = (const float*)d_in[4];
  float* out = (float*)d_out;

  const size_t qkv_elems = (size_t)Bn * Hn * Tn * Dn;  // 4,194,304
  char* ws = (char*)d_ws;
  __bf16* q = (__bf16*)ws;
  __bf16* k = q + qkv_elems;
  __bf16* v = k + qkv_elems;
  __bf16* y = v + qkv_elems;  // [B,T,C] bf16; total ws use = 32 MB

  qkv_gemm<<<dim3(Mtot / 128, (3 * Cn) / 128), 256, 0, stream>>>(
      x, w_attn, b_attn, q, k, v);
  flash_attn<<<dim3(Tn / 64, Bn * Hn), 128, 0, stream>>>(q, k, v, y);
  proj_gemm<<<dim3(Mtot / 128, Cn / 128), 256, 0, stream>>>(y, w_proj, b_proj,
                                                            out);
}
